// FwdAttention_35673998361263
// MI455X (gfx1250) — matrix operand order repaired and checked by reading, not compiled, not run on hardware
//
#include <hip/hip_runtime.h>
#include <math.h>

// ---------------------------------------------------------------------------
// Additive (Bahdanau) attention, fused for MI455X (gfx1250, wave32, WMMA+TDM).
//   B=32, S=4096, CTXDIM=512, QRYDIM=512, ENCDIM=256
// Dominant GEMM (ctx@Wc, 34.4 GFLOP) runs on v_wmma_f32_16x16x32_bf16.
// Wc K-chunks are staged into LDS by the Tensor Data Mover (tensor_load_to_lds)
// with TDM padding replicating the 80B bank-conflict-free row stride.
// ctx (256MB) streams from HBM twice (scores pass + summary pass).
// ---------------------------------------------------------------------------

#define BATCH   32
#define SEQ     4096
#define CTXD    512
#define QRYD    512
#define ENC     256

#define TILE_M  128          // seq rows per workgroup in scores pass
#define KSTEP   32           // K per WMMA (bf16 16x16x32)
#define LDA_STRIDE 40        // padded bf16 row stride (80B, 16B aligned)
#define LDB_STRIDE 40        // must match TDM pad: 64B row + 16B pad = 80B

typedef __attribute__((ext_vector_type(16))) __bf16 v16bf;
typedef __attribute__((ext_vector_type(8)))  float  v8f;
typedef __attribute__((ext_vector_type(4)))  float  v4f;
typedef __attribute__((ext_vector_type(4)))  __bf16 v4bf;

#if __has_builtin(__builtin_amdgcn_tensor_load_to_lds) && \
    __has_builtin(__builtin_amdgcn_s_wait_tensorcnt)
#define USE_TDM 1
#else
#define USE_TDM 0
#endif

typedef __attribute__((ext_vector_type(4))) unsigned int tdm_g0_t;
typedef __attribute__((ext_vector_type(8))) int          tdm_g1_t;
typedef __attribute__((ext_vector_type(4))) int          tdm_g2_t;

union FragCast {
    uint4 u[2];
    v16bf v;
};

// assemble a 16-element bf16 fragment from two 16-byte LDS chunks
static __device__ __forceinline__ v16bf ld_frag(const unsigned short* p0,
                                                const unsigned short* p1) {
    FragCast t;
    t.u[0] = *(const uint4*)p0;
    t.u[1] = *(const uint4*)p1;
    return t.v;
}

union PackBF {          // 16 floats -> 16 bf16 (32B) via packed converts
    v4bf b[4];
    uint4 u[2];
};

// ---------------------------------------------------------------------------
// Kernel 0a: qh[b][e] = qry[b] . Wq[:,e] + b1[e]   (32x256 outputs, trivial)
// ---------------------------------------------------------------------------
__global__ __launch_bounds__(256) void qh_kernel(const float* __restrict__ qry,
                                                 const float* __restrict__ Wq,
                                                 const float* __restrict__ b1,
                                                 float* __restrict__ qh) {
    __shared__ float q[QRYD];
    const int b = blockIdx.x;
    const int t = threadIdx.x;
    q[t]       = qry[b * QRYD + t];
    q[t + 256] = qry[b * QRYD + t + 256];
    __syncthreads();
    float s = b1[t];
    #pragma unroll 8
    for (int k = 0; k < QRYD; ++k)
        s = fmaf(q[k], Wq[k * ENC + t], s);
    qh[b * ENC + t] = s;
}

// ---------------------------------------------------------------------------
// Kernel 0b: WcT[e][k] = bf16(Wc[k][e])   (transposed so B-fragments are
// contiguous 16B runs over K; also the 2D tensor the TDM descriptor walks)
// ---------------------------------------------------------------------------
__global__ __launch_bounds__(256) void wct_kernel(const float* __restrict__ Wc,
                                                  unsigned short* __restrict__ WcT) {
    int idx = blockIdx.x * 256 + threadIdx.x;      // 512*256 elements
    int k = idx / ENC;
    int e = idx % ENC;
    WcT[e * CTXD + k] = __builtin_bit_cast(unsigned short, (__bf16)Wc[k * ENC + e]);
}

// ---------------------------------------------------------------------------
// Kernel 1: fused scores.
//   Per workgroup: 128 seq rows of one batch.
//   H = ctx_tile(128x512) @ Wc(512x256)  via bf16 WMMA, f32 accum,
//   score[s] = tanh(H[s,:] + qh[b,:]) . w2 + b2 + log(mask[b,s])
// 8 waves; wave w owns rows m0 = 16*w; 16 live v8f accumulator tiles per wave.
// B tile staged by the Tensor Data Mover (wave 0 issues, TENSORcnt wait,
// workgroup barrier publishes). A tile staged with packed f32->bf16 converts.
// ---------------------------------------------------------------------------
__global__ __launch_bounds__(256, 1)
void scores_kernel(const float* __restrict__ ctx,
                   const int* __restrict__ mask,
                   const unsigned short* __restrict__ WcT,   // [ENC][CTXD] bf16
                   const float* __restrict__ qh,             // [B][ENC]
                   const float* __restrict__ w2,
                   const float* __restrict__ b2,
                   float* __restrict__ scores_out) {
    __shared__ unsigned short ldsA[TILE_M * LDA_STRIDE];  // 10 KB
    __shared__ unsigned short ldsB[ENC * LDB_STRIDE];     // 20 KB

    const int tid  = threadIdx.x;
    const int b    = blockIdx.x;        // batch
    const int s0   = blockIdx.y * TILE_M;
    const int wave = tid >> 5;          // 0..7
    const int lane = tid & 31;
    const int m0   = wave * 16;
    const int lhalf = lane >> 4;        // 0: lanes 0-15, 1: lanes 16-31
    const int lmod  = lane & 15;

    // Per-lane column constants: lane's column inside each N-chunk is lmod.
    float qv[16], wv[16];
    #pragma unroll
    for (int n = 0; n < 16; ++n) {
        qv[n] = qh[b * ENC + n * 16 + lmod];
        wv[n] = w2[n * 16 + lmod];
    }

    v8f acc[16];
    #pragma unroll
    for (int n = 0; n < 16; ++n)
        #pragma unroll
        for (int r = 0; r < 8; ++r)
            acc[n][r] = 0.0f;

    // cooperative-load thread mapping for A (ctx tile)
    const int arow_ld = tid >> 1;            // 0..127
    const int acol_ld = (tid & 1) * 16;      // 0 or 16

#if USE_TDM
    // LDS byte offset of ldsB (flat-address low 32 bits == LDS offset)
    const unsigned ldsB_off = (unsigned)(unsigned long long)(void*)ldsB;
#endif

    for (int kk = 0; kk < CTXD / KSTEP; ++kk) {
        __syncthreads();   // previous iteration's fragment reads done

        // ---- stage B: WcT[0:256][kk*32 .. +32) -> LDS, padded 80B rows ----
#if USE_TDM
        if (wave == 0) {
            // D# per CDNA5 ISA
            unsigned long long ga =
                (unsigned long long)(const void*)WcT +
                (unsigned long long)(kk * KSTEP * 2);
            tdm_g0_t g0;
            g0.x = 1u;                                   // count=1 (valid D#)
            g0.y = ldsB_off;                             // lds_addr
            g0.z = (unsigned)(ga & 0xFFFFFFFFu);         // global_addr[31:0]
            g0.w = (unsigned)((ga >> 32) & 0x01FFFFFFu)  // global_addr[56:32]
                 | (2u << 30);                           // type=2 ("image")
            tdm_g1_t g1;
            g1[0] = (int)((1u << 16)      // data_size = 2 bytes
                        | (1u << 20)      // pad_enable
                        | (3u << 22)      // pad_interval: every 16 DWORDs (64B row)
                        | (3u << 25));    // pad_amount: 4 DWORDs (16B) -> 80B stride
            g1[1] = (int)(512u << 16);    // tensor_dim0 = 512 (elems)
            g1[2] = (int)(256u << 16);    // tensor_dim1 = 256
            g1[3] = (int)(32u  << 16);    // tile_dim0  = 32
            g1[4] = (int)(256u);          // tile_dim1  = 256, tile_dim2 = 0
            g1[5] = (int)(512u);          // tensor_dim0_stride = 512
            g1[6] = 0;
            g1[7] = 0;
            tdm_g2_t gz = {0, 0, 0, 0};
#if defined(__clang_major__) && (__clang_major__ >= 23)
            tdm_g1_t gz8 = {0, 0, 0, 0, 0, 0, 0, 0};
            __builtin_amdgcn_tensor_load_to_lds(g0, g1, gz, gz, gz8, 0);
#else
            __builtin_amdgcn_tensor_load_to_lds(g0, g1, gz, gz, 0);
#endif
        }
#else
        {
            const unsigned short* src = WcT + (size_t)tid * CTXD + kk * KSTEP;
            uint4 u0 = *(const uint4*)(src);
            uint4 u1 = *(const uint4*)(src + 8);
            unsigned short* dst = ldsB + tid * LDB_STRIDE;
            *(uint4*)(dst)     = u0;
            *(uint4*)(dst + 8) = u1;
        }
#endif

        // ---- stage A: ctx[b][s0+row][kk*32 + c] -> bf16 LDS (packed cvt) ----
        {
            const v4f* src = (const v4f*)(ctx + ((size_t)b * SEQ + s0 + arow_ld) * CTXD
                                              + kk * KSTEP + acol_ld);
            if (kk + 1 < CTXD / KSTEP)
                __builtin_prefetch((const float*)src + KSTEP, 0, 1);
            PackBF pk;
            pk.b[0] = __builtin_convertvector(src[0], v4bf);
            pk.b[1] = __builtin_convertvector(src[1], v4bf);
            pk.b[2] = __builtin_convertvector(src[2], v4bf);
            pk.b[3] = __builtin_convertvector(src[3], v4bf);
            unsigned short* dst = ldsA + arow_ld * LDA_STRIDE + acol_ld;
            *(uint4*)(dst)     = pk.u[0];   // row*80 + {0,32}: 16B aligned
            *(uint4*)(dst + 8) = pk.u[1];
        }

#if USE_TDM
        if (wave == 0)
            __builtin_amdgcn_s_wait_tensorcnt(0);   // TDM tile landed in LDS
#endif
        __syncthreads();

        // ---- A fragment (16x32 bf16, ISA layout): lane = row m0+lmod,
        //      K chunks {8*lhalf..} and {16+8*lhalf..} ----
        const unsigned short* pa = ldsA + (m0 + lmod) * LDA_STRIDE + lhalf * 8;
        v16bf afrag = ld_frag(pa, pa + 16);

        // ---- 16 N-chunks: B fragment K chunks {8*lhalf..} and {16+8*lhalf..} ----
        #pragma unroll
        for (int n = 0; n < 16; ++n) {
            const unsigned short* pb =
                ldsB + (n * 16 + lmod) * LDB_STRIDE + lhalf * 8;
            v16bf bfrag = ld_frag(pb, pb + 16);
            acc[n] = __builtin_amdgcn_wmma_f32_16x16x32_bf16(
                false, afrag, false, bfrag, (short)0, acc[n], false, false);
        }
    }

    // ---- epilogue: score_part[r] = sum_n tanh(acc + qh) * w2 over this
    //      lane's 16 columns, rows r (+8 if lhalf) ----
    float sp[8];
    #pragma unroll
    for (int r = 0; r < 8; ++r) sp[r] = 0.0f;
    #pragma unroll
    for (int n = 0; n < 16; ++n) {
        #pragma unroll
        for (int r = 0; r < 8; ++r) {
            float h = tanhf(acc[n][r] + qv[n]);
            sp[r] = fmaf(h, wv[n], sp[r]);
        }
    }
    // reduce over the 16 lanes of each half-wave (columns lmod = 0..15)
    #pragma unroll
    for (int off = 8; off >= 1; off >>= 1)
        #pragma unroll
        for (int r = 0; r < 8; ++r)
            sp[r] += __shfl_xor(sp[r], off, 16);

    if (lmod == 0) {
        const float bias2 = b2[0];
        const int rbase = s0 + m0 + lhalf * 8;
        #pragma unroll
        for (int r = 0; r < 8; ++r) {
            int s = rbase + r;
            float sc = sp[r] + bias2 + logf((float)mask[b * SEQ + s]);
            scores_out[(size_t)b * SEQ + s] = sc;
        }
    }
}

// ---------------------------------------------------------------------------
// Kernel 2: softmax over S per batch row; also zero-inits summary region.
// ---------------------------------------------------------------------------
__global__ __launch_bounds__(256) void softmax_kernel(const float* __restrict__ scores,
                                                      float* __restrict__ alphas,
                                                      float* __restrict__ summary) {
    __shared__ float red[256];
    const int b = blockIdx.x;
    const int t = threadIdx.x;
    float vals[SEQ / 256];
    float m = -INFINITY;
    #pragma unroll
    for (int i = 0; i < SEQ / 256; ++i) {
        vals[i] = scores[(size_t)b * SEQ + t + i * 256];
        m = fmaxf(m, vals[i]);
    }
    red[t] = m; __syncthreads();
    for (int s = 128; s > 0; s >>= 1) {
        if (t < s) red[t] = fmaxf(red[t], red[t + s]);
        __syncthreads();
    }
    m = red[0]; __syncthreads();
    float sum = 0.0f;
    #pragma unroll
    for (int i = 0; i < SEQ / 256; ++i) {
        vals[i] = __expf(vals[i] - m);   // exp(-inf) = 0 handles masked slots
        sum += vals[i];
    }
    red[t] = sum; __syncthreads();
    for (int s = 128; s > 0; s >>= 1) {
        if (t < s) red[t] += red[t + s];
        __syncthreads();
    }
    const float inv = 1.0f / red[0];
    #pragma unroll
    for (int i = 0; i < SEQ / 256; ++i)
        alphas[(size_t)b * SEQ + t + i * 256] = vals[i] * inv;

    // zero summary (atomically accumulated by next kernel, ordered by stream)
    summary[b * CTXD + t] = 0.0f;
    summary[b * CTXD + t + 256] = 0.0f;
}

// ---------------------------------------------------------------------------
// Kernel 3: summary[b][d] = sum_s alphas[b][s] * ctx[b][s][d]
// grid (B, S/512): each block accumulates a 512-row partial via atomicAdd.
// Second (unavoidable) HBM pass over ctx; fully coalesced f32 reads.
// ---------------------------------------------------------------------------
__global__ __launch_bounds__(256) void summary_kernel(const float* __restrict__ alphas,
                                                      const float* __restrict__ ctx,
                                                      float* __restrict__ summary) {
    __shared__ float a[512];
    const int b  = blockIdx.x;
    const int s0 = blockIdx.y * 512;
    const int t  = threadIdx.x;
    a[t]       = alphas[(size_t)b * SEQ + s0 + t];
    a[t + 256] = alphas[(size_t)b * SEQ + s0 + t + 256];
    __syncthreads();
    float acc0 = 0.0f, acc1 = 0.0f;
    for (int s = 0; s < 512; ++s) {
        const float al = a[s];
        const float* row = ctx + ((size_t)b * SEQ + s0 + s) * CTXD;
        acc0 = fmaf(al, row[t], acc0);
        acc1 = fmaf(al, row[t + 256], acc1);
    }
    atomicAdd(&summary[b * CTXD + t], acc0);
    atomicAdd(&summary[b * CTXD + t + 256], acc1);
}

// ---------------------------------------------------------------------------
extern "C" void kernel_launch(void* const* d_in, const int* in_sizes, int n_in,
                              void* d_out, int out_size, void* d_ws, size_t ws_size,
                              hipStream_t stream) {
    const float* qry  = (const float*)d_in[0];   // (32,512)
    const float* ctx  = (const float*)d_in[1];   // (32,4096,512)
    const int*   mask = (const int*)d_in[2];     // (32,4096)
    const float* Wc   = (const float*)d_in[3];   // (512,256)
    const float* Wq   = (const float*)d_in[4];   // (512,256)
    const float* b1   = (const float*)d_in[5];   // (256,)
    const float* w2   = (const float*)d_in[6];   // (256,)
    const float* b2   = (const float*)d_in[7];   // scalar

    float* out     = (float*)d_out;
    float* alphas  = out;                                // 32*4096
    float* summary = out + BATCH * SEQ;                  // 32*512
    float* scores  = out + BATCH * SEQ + BATCH * CTXD;   // 32*4096

    float*          qh  = (float*)d_ws;                              // 32 KB
    unsigned short* WcT = (unsigned short*)((char*)d_ws + BATCH * ENC * sizeof(float)); // 256 KB

    qh_kernel <<<BATCH, 256, 0, stream>>>(qry, Wq, b1, qh);
    wct_kernel<<<(CTXD * ENC) / 256, 256, 0, stream>>>(Wc, WcT);
    scores_kernel<<<dim3(BATCH, SEQ / TILE_M), 256, 0, stream>>>(
        ctx, mask, WcT, qh, w2, b2, scores);
    softmax_kernel<<<BATCH, 256, 0, stream>>>(scores, alphas, summary);
    summary_kernel<<<dim3(BATCH, SEQ / 512), 256, 0, stream>>>(alphas, ctx, summary);
}
